// MyModel_11398843203601
// MI455X (gfx1250) — matrix operand order repaired and checked by reading, not compiled, not run on hardware
//
#include <hip/hip_runtime.h>
#include <math.h>

typedef __attribute__((ext_vector_type(16))) _Float16 v16h;
typedef __attribute__((ext_vector_type(8)))  _Float16 v8h;
typedef __attribute__((ext_vector_type(8)))  float    v8f;

union AFrag { v16h v; v8h q[2]; _Float16 h[16]; };
union CFrag { v8f  v; float f[8]; };

__device__ inline v8f wmma16x16x32(v16h a, v16h b, v8f c) {
  return __builtin_amdgcn_wmma_f32_16x16x32_f16(false, a, false, b, (short)0, c, false, false);
}

// A fragment (16x32 f16) from LDS row-major tile, lda in halves (multiple of 8).
// Lane l: row m=l&15, K = kb..kb+7 and 16+kb..16+kb+7, kb=8*(l>=16).
__device__ inline v16h lds_a_frag(const _Float16* a0, int lda) {
  int lane = threadIdx.x & 31;
  const _Float16* r = a0 + (lane & 15) * lda + ((lane >> 4) << 3);
  AFrag f;
  f.q[0] = *(const v8h*)(r);
  f.q[1] = *(const v8h*)(r + 16);
  return f.v;
}

// B fragment (32x16 f16) from LDS stored N-MAJOR: Bn[n*ldk + k], ldk mult of 8.
// Lane l: col n=l&15, K = kb..kb+7 and 16+kb..16+kb+7, kb=8*(l>=16).
__device__ inline v16h lds_b_frag_nm(const _Float16* b0, int ldk) {
  int lane = threadIdx.x & 31;
  const _Float16* r = b0 + (lane & 15) * ldk + ((lane >> 4) << 3);
  AFrag f;
  f.q[0] = *(const v8h*)(r);
  f.q[1] = *(const v8h*)(r + 16);
  return f.v;
}

__device__ inline void copy8h(_Float16* dst, const _Float16* src) {
#pragma unroll
  for (int d = 0; d < 8; ++d) dst[d] = src[d];
}

// async global<->LDS 16-byte transfers (ASYNCcnt-tracked)
__device__ inline void async_load_b128(unsigned ldsoff, const void* gaddr) {
  asm volatile("global_load_async_to_lds_b128 %0, %1, off"
               :: "v"(ldsoff), "v"((unsigned long long)gaddr) : "memory");
}
__device__ inline void async_store_b128(void* gaddr, unsigned ldsoff) {
  asm volatile("global_store_async_from_lds_b128 %0, %1, off"
               :: "v"((unsigned long long)gaddr), "v"(ldsoff) : "memory");
}
__device__ inline void wait_async0() {
  asm volatile("s_wait_asynccnt 0" ::: "memory");
}

// ---------------------------------------------------------------------------
// K1: block-linear (permutation composed away) -> img f16 [B,3,32,32] NCHW
// out(b,c,h,w) = sum_k W[(h%4*4+w%4)*3+c, k] * xb[b,(h/4)*8+(w/4), k]
// One WG (128 blocks) produces exactly 2 contiguous images -> LDS-staged
// output drained with coalesced async b128 stores.
// ---------------------------------------------------------------------------
__global__ __launch_bounds__(256) void k1_blocklin(
    const float* __restrict__ x, const float* __restrict__ W,
    _Float16* __restrict__ img) {
  __shared__ __align__(16) _Float16 As[128 * 64];   // K padded 48->64
  __shared__ __align__(16) _Float16 Bs[48 * 64];    // N-major: Bs[j*64+k]=W[j,k]
  __shared__ __align__(16) _Float16 outs[2 * 3072]; // 2 staged NCHW images
  const int tid = threadIdx.x;
  const int base = blockIdx.x * 128;

  for (int idx = tid; idx < 48 * 64; idx += 256) {
    int j = idx >> 6, k = idx & 63;
    Bs[idx] = (k < 48) ? (_Float16)W[j * 48 + k] : (_Float16)0.0f;
  }
  // A: each segment = 12 contiguous floats (one 4x1x3 strip of a block)
  for (int seg = tid; seg < 512; seg += 256) {
    int row = seg >> 2, p = seg & 3;
    int gblk = base + row;
    int b = gblk >> 6, blk = gblk & 63;
    const float* src = x + (((size_t)b * 32 + (blk >> 3) * 4 + p) * 32 + (blk & 7) * 4) * 3;
    _Float16* dst = As + row * 64 + p * 12;
#pragma unroll
    for (int j = 0; j < 12; ++j) dst[j] = (_Float16)src[j];
  }
  for (int idx = tid; idx < 128 * 16; idx += 256)
    As[(idx >> 4) * 64 + 48 + (idx & 15)] = (_Float16)0.0f;
  __syncthreads();

  const int wv = tid >> 5, lane = tid & 31;
  const _Float16* arow = As + wv * 16 * 64;
#pragma unroll
  for (int jt = 0; jt < 3; ++jt) {
    int j0 = jt * 16;
    CFrag cf; cf.v = (v8f)0.0f;
#pragma unroll
    for (int kc = 0; kc < 2; ++kc) {
      v16h a = lds_a_frag(arow + kc * 32, 64);
      v16h b = lds_b_frag_nm(Bs + j0 * 64 + kc * 32, 64);
      cf.v = wmma16x16x32(a, b, cf.v);
    }
    int jcol = j0 + (lane & 15);
    int pq = jcol / 3, c = jcol % 3;
    int hoff = pq >> 2, woff = pq & 3;
#pragma unroll
    for (int r = 0; r < 8; ++r) {
      int crow = ((lane >> 4) << 3) + r;
      int lblk = wv * 16 + crow;              // 0..127 within this WG
      int bl = lblk >> 6, blk = lblk & 63;
      int h = (blk >> 3) * 4 + hoff, wq = (blk & 7) * 4 + woff;
      outs[bl * 3072 + (c << 10) + (h << 5) + wq] = (_Float16)cf.f[r];
    }
  }
  __syncthreads();
  _Float16* gdst = img + (size_t)base * 48;   // = 2 images, 6144 halves
  for (int idx = tid; idx < 768; idx += 256)
    async_store_b128(gdst + idx * 8, (unsigned)(size_t)(outs + idx * 8));
  wait_async0();
}

// ---------------------------------------------------------------------------
// K2: conv1 (3->10,5x5)+bias+maxpool2+relu -> pool1 f16 [B,10,14,14]
// K reordered: k' = c*40 + dy*8 + dx (dx padded 5->8), Kpad = 128 (4 chunks)
// ---------------------------------------------------------------------------
__device__ inline v16h a_frag_conv1(const _Float16* imgs, int t, int kc) {
  int lane = threadIdx.x & 31;
  int m = t * 16 + (lane & 15);
  int oy = m / 28, ox = m % 28;
  int kb = (lane >> 4) << 3;
  AFrag f;
#pragma unroll
  for (int e8 = 0; e8 < 2; ++e8) {
    int Kp = kc * 32 + e8 * 16 + kb;           // multiple of 8
    if (Kp < 120) {
      int c = Kp / 40, r = Kp % 40, dy = r >> 3;
      copy8h(&f.h[e8 * 8], imgs + (c * 32 + oy + dy) * 32 + ox);
    } else {
#pragma unroll
      for (int d = 0; d < 8; ++d) f.h[e8 * 8 + d] = (_Float16)0.0f;
    }
  }
  return f.v;
}

__global__ __launch_bounds__(256) void k2_conv1(
    const _Float16* __restrict__ img, const float* __restrict__ w1,
    const float* __restrict__ b1, _Float16* __restrict__ pool1) {
  __shared__ __align__(16) _Float16 imgs[3072 + 16];   // padded finite tail
  __shared__ __align__(16) _Float16 B1s[16 * 128];     // N-major, K' order
  __shared__ __align__(16) _Float16 pbuf[1960];        // staged pooled output
  __shared__ float conv1s[10 * 784];
  __shared__ float b1s[16];
  const int tid = threadIdx.x, b = blockIdx.x;

  for (int idx = tid; idx < 384; idx += 256)           // 384 x 16B image stage
    async_load_b128((unsigned)(size_t)(imgs + idx * 8), img + (size_t)b * 3072 + idx * 8);
  for (int idx = tid; idx < 16 * 128; idx += 256) {
    int n = idx >> 7, kp = idx & 127;
    int c = kp / 40, r = kp % 40, dy = r >> 3, dx = r & 7;
    _Float16 v = (_Float16)0.0f;
    if (n < 10 && kp < 120 && dx < 5) v = (_Float16)w1[n * 75 + c * 25 + dy * 5 + dx];
    B1s[idx] = v;
  }
  if (tid < 16) {
    b1s[tid] = (tid < 10) ? b1[tid] : 0.0f;
    imgs[3072 + tid] = (_Float16)0.0f;
  }
  wait_async0();
  __syncthreads();

  const int wv = tid >> 5, lane = tid & 31;
  v16h bf[4];
#pragma unroll
  for (int kc = 0; kc < 4; ++kc) bf[kc] = lds_b_frag_nm(B1s + kc * 32, 128);

  for (int t = wv; t < 49; t += 8) {
    CFrag cf; cf.v = (v8f)0.0f;
#pragma unroll
    for (int kc = 0; kc < 4; ++kc)
      cf.v = wmma16x16x32(a_frag_conv1(imgs, t, kc), bf[kc], cf.v);
    int n = lane & 15;
    if (n < 10) {
#pragma unroll
      for (int r = 0; r < 8; ++r) {
        int m = t * 16 + ((lane >> 4) << 3) + r;
        conv1s[n * 784 + m] = cf.f[r] + b1s[n];
      }
    }
  }
  __syncthreads();
  for (int idx = tid; idx < 10 * 196; idx += 256) {
    int ch = idx / 196, rem = idx % 196;
    int py = rem / 14, px = rem % 14;
    const float* p = conv1s + ch * 784 + (2 * py) * 28 + 2 * px;
    float v = fmaxf(fmaxf(p[0], p[1]), fmaxf(p[28], p[29]));
    pbuf[idx] = (_Float16)fmaxf(v, 0.0f);
  }
  __syncthreads();
  for (int idx = tid; idx < 245; idx += 256)           // 245 x 16B drain
    async_store_b128(pool1 + (size_t)b * 1960 + idx * 8, (unsigned)(size_t)(pbuf + idx * 8));
  wait_async0();
}

// ---------------------------------------------------------------------------
// K3: conv2 (10->20,5x5)+bias+maxpool2+relu -> fcin f16 [B,512] (500 + zero pad)
// K reordered: k' = ch*40 + dy*8 + dx, Kpad = 416 (13 chunks), N 20->32
// ---------------------------------------------------------------------------
__device__ inline v16h a_frag_conv2(const _Float16* p1s, int t, int kc) {
  int lane = threadIdx.x & 31;
  int m = t * 16 + (lane & 15);
  int kb = (lane >> 4) << 3;
  AFrag f;
  if (m < 100) {
    int oy = m / 10, ox = m % 10;
#pragma unroll
    for (int e8 = 0; e8 < 2; ++e8) {
      int Kp = kc * 32 + e8 * 16 + kb;
      if (Kp < 400) {
        int ch = Kp / 40, r = Kp % 40, dy = r >> 3;
        copy8h(&f.h[e8 * 8], p1s + (ch * 14 + oy + dy) * 14 + ox);
      } else {
#pragma unroll
        for (int d = 0; d < 8; ++d) f.h[e8 * 8 + d] = (_Float16)0.0f;
      }
    }
  } else {
#pragma unroll
    for (int e = 0; e < 16; ++e) f.h[e] = (_Float16)0.0f;
  }
  return f.v;
}

__global__ __launch_bounds__(256) void k3_conv2(
    const _Float16* __restrict__ pool1, const float* __restrict__ w2,
    const float* __restrict__ b2, _Float16* __restrict__ fcin) {
  __shared__ __align__(16) _Float16 p1s[1960 + 16];
  __shared__ __align__(16) _Float16 B2s[32 * 416];   // N-major, K' order
  __shared__ __align__(16) _Float16 fcbuf[512];
  __shared__ float conv2s[20 * 100];
  __shared__ float b2s[32];
  const int tid = threadIdx.x, b = blockIdx.x;

  for (int idx = tid; idx < 245; idx += 256)   // 245 x 16B = 1960 halves
    async_load_b128((unsigned)(size_t)(p1s + idx * 8), pool1 + (size_t)b * 1960 + idx * 8);
  for (int idx = tid; idx < 32 * 416; idx += 256) {
    int n = idx / 416, kp = idx % 416;
    int ch = kp / 40, r = kp % 40, dy = r >> 3, dx = r & 7;
    _Float16 v = (_Float16)0.0f;
    if (n < 20 && kp < 400 && dx < 5) v = (_Float16)w2[n * 250 + ch * 25 + dy * 5 + dx];
    B2s[idx] = v;
  }
  if (tid < 32) b2s[tid] = (tid < 20) ? b2[tid] : 0.0f;
  if (tid < 16) p1s[1960 + tid] = (_Float16)0.0f;
  wait_async0();
  __syncthreads();

  const int wv = tid >> 5, lane = tid & 31;
  for (int t = wv; t < 7; t += 8) {
#pragma unroll
    for (int nt = 0; nt < 2; ++nt) {
      CFrag cf; cf.v = (v8f)0.0f;
#pragma unroll
      for (int kc = 0; kc < 13; ++kc) {
        v16h a = a_frag_conv2(p1s, t, kc);
        v16h bb = lds_b_frag_nm(B2s + (nt * 16) * 416 + kc * 32, 416);
        cf.v = wmma16x16x32(a, bb, cf.v);
      }
      int n = nt * 16 + (lane & 15);
      if (n < 20) {
#pragma unroll
        for (int r = 0; r < 8; ++r) {
          int m = t * 16 + ((lane >> 4) << 3) + r;
          if (m < 100) conv2s[n * 100 + m] = cf.f[r] + b2s[n];
        }
      }
    }
  }
  __syncthreads();
  for (int idx = tid; idx < 512; idx += 256) {
    _Float16 v = (_Float16)0.0f;
    if (idx < 500) {
      int ch = idx / 25, rem = idx % 25;
      int py = rem / 5, px = rem % 5;
      const float* p = conv2s + ch * 100 + (2 * py) * 10 + 2 * px;
      float m4 = fmaxf(fmaxf(p[0], p[1]), fmaxf(p[10], p[11]));
      v = (_Float16)fmaxf(m4, 0.0f);
    }
    fcbuf[idx] = v;
  }
  __syncthreads();
  if (tid < 64)                                 // 64 x 16B drain
    async_store_b128(fcin + (size_t)b * 512 + tid * 8, (unsigned)(size_t)(fcbuf + tid * 8));
  wait_async0();
}

// ---------------------------------------------------------------------------
// K4: fc1(500->50)+relu, fc2(50->10), log_softmax -> out f32 [B,10]
// 128 images/WG (16 per wave); fc1 weights staged N-major in two 32-col passes
// ---------------------------------------------------------------------------
__device__ inline v16h a_frag_fc(const _Float16* __restrict__ fcin, int row0, int kc) {
  int lane = threadIdx.x & 31;
  const _Float16* r = fcin + (size_t)(row0 + (lane & 15)) * 512 + kc * 32 + ((lane >> 4) << 3);
  AFrag f;
  f.q[0] = *(const v8h*)(r);
  f.q[1] = *(const v8h*)(r + 16);
  return f.v;
}

__global__ __launch_bounds__(256) void k4_fc(
    const _Float16* __restrict__ fcin,
    const float* __restrict__ f1w, const float* __restrict__ f1b,
    const float* __restrict__ f2w, const float* __restrict__ f2b,
    float* __restrict__ out) {
  __shared__ __align__(16) _Float16 W1s[32 * 512];    // N-major pass tile
  __shared__ __align__(16) _Float16 W2s[16 * 64];     // N-major fc2^ (K 50->64)
  __shared__ __align__(16) _Float16 h1s[8 * 16 * 64];
  __shared__ float lg[8 * 16 * 16];
  __shared__ float sb1[64];
  __shared__ float sb2[16];
  const int tid = threadIdx.x, wv = tid >> 5, lane = tid & 31;
  const int row0 = blockIdx.x * 128 + wv * 16;

  for (int idx = tid; idx < 16 * 64; idx += 256) {
    int n = idx >> 6, k = idx & 63;
    W2s[idx] = (n < 10 && k < 50) ? (_Float16)f2w[n * 50 + k] : (_Float16)0.0f;
  }
  if (tid < 64) sb1[tid] = (tid < 50) ? f1b[tid] : 0.0f;
  if (tid < 16) sb2[tid] = (tid < 10) ? f2b[tid] : 0.0f;

  for (int pass = 0; pass < 2; ++pass) {
    int n0 = pass * 32;
    __syncthreads();
    for (int idx = tid; idx < 32 * 512; idx += 256) {
      int n = n0 + (idx >> 9), k = idx & 511;
      W1s[idx] = (n < 50 && k < 500) ? (_Float16)f1w[n * 500 + k] : (_Float16)0.0f;
    }
    __syncthreads();
    CFrag acc0, acc1; acc0.v = (v8f)0.0f; acc1.v = (v8f)0.0f;
#pragma unroll
    for (int kc = 0; kc < 16; ++kc) {
      v16h a = a_frag_fc(fcin, row0, kc);
      acc0.v = wmma16x16x32(a, lds_b_frag_nm(W1s + kc * 32, 512), acc0.v);
      acc1.v = wmma16x16x32(a, lds_b_frag_nm(W1s + 16 * 512 + kc * 32, 512), acc1.v);
    }
#pragma unroll
    for (int r = 0; r < 8; ++r) {
      int crow = ((lane >> 4) << 3) + r;
      int na = n0 + (lane & 15), nb = n0 + 16 + (lane & 15);
      h1s[wv * 1024 + crow * 64 + na] = (_Float16)fmaxf(acc0.f[r] + sb1[na], 0.0f);
      h1s[wv * 1024 + crow * 64 + nb] = (_Float16)fmaxf(acc1.f[r] + sb1[nb], 0.0f);
    }
  }
  __syncthreads();

  CFrag c2; c2.v = (v8f)0.0f;
#pragma unroll
  for (int kc = 0; kc < 2; ++kc) {
    v16h a = lds_a_frag(h1s + wv * 1024 + kc * 32, 64);
    c2.v = wmma16x16x32(a, lds_b_frag_nm(W2s + kc * 32, 64), c2.v);
  }
#pragma unroll
  for (int r = 0; r < 8; ++r) {
    int crow = ((lane >> 4) << 3) + r;
    lg[wv * 256 + crow * 16 + (lane & 15)] = c2.f[r] + sb2[lane & 15];
  }
  __syncthreads();

  if (lane < 16) {
    const float* row = lg + wv * 256 + lane * 16;
    float mx = row[0];
#pragma unroll
    for (int n = 1; n < 10; ++n) mx = fmaxf(mx, row[n]);
    float s = 0.0f;
#pragma unroll
    for (int n = 0; n < 10; ++n) s += __expf(row[n] - mx);
    float lse = mx + __logf(s);
    float* o = out + (size_t)(row0 + lane) * 10;
#pragma unroll
    for (int n = 0; n < 10; ++n) o[n] = row[n] - lse;
  }
}

// ---------------------------------------------------------------------------
extern "C" void kernel_launch(void* const* d_in, const int* in_sizes, int n_in,
                              void* d_out, int out_size, void* d_ws, size_t ws_size,
                              hipStream_t stream) {
  const float* x   = (const float*)d_in[0];
  const float* W   = (const float*)d_in[1];
  const float* c1w = (const float*)d_in[2];
  const float* c1b = (const float*)d_in[3];
  const float* c2w = (const float*)d_in[4];
  const float* c2b = (const float*)d_in[5];
  const float* f1w = (const float*)d_in[6];
  const float* f1b = (const float*)d_in[7];
  const float* f2w = (const float*)d_in[8];
  const float* f2b = (const float*)d_in[9];
  float* out = (float*)d_out;
  const int B = in_sizes[0] / (32 * 32 * 3);

  auto align256 = [](size_t v) { return (v + 255) & ~(size_t)255; };
  char* ws = (char*)d_ws;
  _Float16* img   = (_Float16*)ws;
  size_t off = align256((size_t)B * 3072 * sizeof(_Float16));
  _Float16* pool1 = (_Float16*)(ws + off);
  off += align256((size_t)B * 1960 * sizeof(_Float16));
  _Float16* fcin  = (_Float16*)(ws + off);

  k1_blocklin<<<(B * 64) / 128, 256, 0, stream>>>(x, W, img);
  k2_conv1  <<<B, 256, 0, stream>>>(img, c1w, c1b, pool1);
  k3_conv2  <<<B, 256, 0, stream>>>(pool1, c2w, c2b, fcin);
  k4_fc     <<<B / 128, 256, 0, stream>>>(fcin, f1w, f1b, f2w, f2b, out);
}
